// ACTHybridTRM_41480794145079
// MI455X (gfx1250) — matrix operand order repaired and checked by reading, not compiled, not run on hardware
//
#include <hip/hip_runtime.h>
#include <hip/hip_bf16.h>
#include <math.h>

typedef _Float16 h16;
typedef __attribute__((ext_vector_type(16))) _Float16 v16h;
typedef __attribute__((ext_vector_type(8)))  float    v8f;
typedef __attribute__((ext_vector_type(4)))  unsigned int v4u;

union F16F { v16h v; v4u u[2]; };
union H24  { v4u u[3]; h16 h[24]; };   // one attention row (24 halves, 16B aligned)
union R48  { v4u u[6]; h16 h[48]; };
union R64  { v4u u[8]; h16 h[64]; };

// ---------------- fragment loaders (CDNA5 WMMA 16x16x32 f16 layouts) -------
// A-matrix 16x32: lane m=lane&15 holds row m; lanes<16 K={0..7,16..23},
// lanes>=16 K={8..15,24..31}  (two aligned 16B runs)
__device__ __forceinline__ v16h frag_A(const h16* base, int ld, int row0, int k0) {
  const int lane = threadIdx.x & 31;
  const h16* r = base + (size_t)(row0 + (lane & 15)) * ld + k0 + ((lane & 16) ? 8 : 0);
  F16F f;
  f.u[0] = *(const v4u*)(r);
  f.u[1] = *(const v4u*)(r + 16);
  return f.v;
}
// B-matrix 32x16 from row-major weights W[N][ld] (computing X @ W^T):
// lane n=lane&15 holds col n; lanes<16 K={0..7,16..23}, lanes>=16 K={8..15,24..31}
__device__ __forceinline__ v16h frag_B(const h16* W, int ld, int n0, int k0) {
  const int lane = threadIdx.x & 31;
  const h16* r = W + (size_t)(n0 + (lane & 15)) * ld + k0 + ((lane & 16) ? 8 : 0);
  F16F f;
  f.u[0] = *(const v4u*)(r);
  f.u[1] = *(const v4u*)(r + 16);
  return f.v;
}

__device__ __forceinline__ v8f wmma_f16(v16h a, v16h b, v8f c) {
  return __builtin_amdgcn_wmma_f32_16x16x32_f16(false, a, false, b, (short)0, c, false, false);
}

// reduce across the 16 lanes of a half-wave (C-fragment row statistics)
__device__ __forceinline__ float redhalf(float v) {
  v += __shfl_xor(v, 1); v += __shfl_xor(v, 2);
  v += __shfl_xor(v, 4); v += __shfl_xor(v, 8);
  return v;
}

__device__ __forceinline__ float gelu_f(float x) {
  return 0.5f * x * (1.f + erff(x * 0.70710678118654752f));
}

// ---------------- workspace layout (element offsets in h16) ----------------
// all weight pads are zero-filled => GEMMs over padded K/N are exact
constexpr size_t OFF_M2VW   = 0;        // [48][224]
constexpr size_t OFF_SAINW  = 10752;    // [144][64]
constexpr size_t OFF_SAOUTW = 19968;    // [48][64]
constexpr size_t OFF_FFW1   = 23040;    // [96][64]
constexpr size_t OFF_FFW2   = 29184;    // [48][96]
constexpr size_t OFF_CAVW   = 33792;    // [48][64]  (rows 96..143 of ca.in_w)
constexpr size_t OFF_CAOUTW = 36864;    // [48][64]
constexpr size_t OFF_POOLW  = 39936;    // [64][64]
constexpr size_t OFF_ZW1    = 44032;    // [112][192]
constexpr size_t OFF_ZW2    = 65536;    // [64][128]
constexpr size_t OFF_YW1    = 73728;    // [112][128]
constexpr size_t OFF_YW2    = 88064;    // [64][128]
constexpr size_t OFF_H1W    = 96256;    // [32][64]  halt.W1 (exact fit)
constexpr size_t OFF_XP     = 98304;    // [B][64] f16

// ---------------- prep: fp32 -> f16 with zero padding ----------------------
__global__ void cvt_pad(const float* __restrict__ src, h16* __restrict__ dst,
                        int N, int K, int Np, int Kp, int rowOff, int srcLd) {
  const int i = blockIdx.x * blockDim.x + threadIdx.x;
  const int tot = Np * Kp;
  if (i >= tot) return;
  const int n = i / Kp, k = i - n * Kp;
  dst[i] = (n < N && k < K) ? (h16)src[(size_t)(rowOff + n) * srcLd + k] : (h16)0;
}

// ---------------- encoder ---------------------------------------------------
struct EncP {
  const float* x;
  const float *tokW, *tokB, *tokG, *tokBeta;
  const float *m2vB, *m2vG, *m2vBeta;
  const float *sa_inb, *sa_outb, *san_g, *san_b;
  const float *ff_b1, *ff_b2, *safn_g, *safn_b;
  const float *ca_inb, *ca_outb, *can_g, *can_b;
  const float *pool_b, *pool_g, *pool_beta;
  const h16 *m2vW, *sainW, *saoutW, *ffW1, *ffW2, *cavW, *caoutW, *poolW;
  h16* xp;
};

constexpr unsigned ENC_LDS_BYTES =
    (22528u + 50688u + 1024u + 1024u + 1024u + 3584u) * 2u + 768u * 4u; // 162816

__global__ __launch_bounds__(256)
void enc_kernel(EncP p) {
  extern __shared__ __align__(16) char smem[];
  h16* tok    = (h16*)smem;            // [352][64]  token activations (f16, K-pad 64)
  h16* qkv    = tok + 22528;           // [352][144] qkv / [352][96] ff-hidden
  h16* ctxh   = qkv + 50688;           // [16][64]
  h16* pooled = ctxh + 1024;           // [16][64]
  h16* vbuf   = pooled + 1024;         // [16][64]
  h16* m2vA   = vbuf + 1024;           // [16][224]
  float* fsc  = (float*)(m2vA + 3584); // 768 floats: tok params, later cross-attn out

  const int tid  = threadIdx.x;
  const int lane = tid & 31;
  const int wave = tid >> 5;
  const int cl   = lane & 15;
  const int mb   = (lane & 16) ? 8 : 0;
  const int e0   = blockIdx.x << 4;

  // zero pads + stage token-proj params into LDS
  for (int i = tid; i < 1024; i += 256) { ctxh[i] = (h16)0; pooled[i] = (h16)0; vbuf[i] = (h16)0; }
  for (int i = tid; i < 288; i += 256) fsc[i] = p.tokW[i];
  if (tid < 48) { fsc[288+tid] = p.tokB[tid]; fsc[336+tid] = p.tokG[tid]; fsc[384+tid] = p.tokBeta[tid]; }
  __syncthreads();

  // S1: token projection (K=6, VALU) + LN + gelu -> tok f16
  for (int r = tid; r < 352; r += 256) {
    const int e = r / 22, t = r - e * 22;
    const float* mg = p.x + (size_t)(e0 + e) * 332 + t * 6;
    float o[48]; float s1 = 0.f, s2 = 0.f;
    #pragma unroll
    for (int c = 0; c < 48; ++c) {
      float a = fsc[288 + c];
      #pragma unroll
      for (int s = 0; s < 6; ++s) a += mg[s] * fsc[c * 6 + s];
      o[c] = a; s1 += a; s2 += a * a;
    }
    const float mu = s1 * (1.f/48.f);
    const float rs = rsqrtf(s2 * (1.f/48.f) - mu * mu + 1e-5f);
    R64 out;
    #pragma unroll
    for (int c = 0; c < 48; ++c) out.h[c] = (h16)gelu_f((o[c]-mu)*rs*fsc[336+c] + fsc[384+c]);
    #pragma unroll
    for (int c = 48; c < 64; ++c) out.h[c] = (h16)0;
    h16* tr = tok + r * 64;
    #pragma unroll
    for (int j = 0; j < 8; ++j) *(v4u*)(tr + 8 * j) = out.u[j];
  }
  // S2: stage m2v inputs as f16 A-tile (K-pad 224)
  for (int i = tid; i < 16 * 224; i += 256) {
    const int rr = i / 224, cc = i - rr * 224;
    m2vA[i] = (cc < 200) ? (h16)p.x[(size_t)(e0 + rr) * 332 + 132 + cc] : (h16)0;
  }
  __syncthreads();

  // S3: qkv = tok @ in_w^T + in_b  (WMMA, M=352,N=144,K=64)
  for (int mt = wave; mt < 22; mt += 8) {
    const v16h a0 = frag_A(tok, 64, mt * 16, 0);
    const v16h a1 = frag_A(tok, 64, mt * 16, 32);
    #pragma unroll
    for (int nt = 0; nt < 9; ++nt) {
      const int c = nt * 16 + cl;
      const float bv = p.sa_inb[c];
      v8f acc;
      #pragma unroll
      for (int r = 0; r < 8; ++r) acc[r] = bv;
      acc = wmma_f16(a0, frag_B(p.sainW, 64, nt * 16, 0), acc);
      acc = wmma_f16(a1, frag_B(p.sainW, 64, nt * 16, 32), acc);
      #pragma unroll
      for (int r = 0; r < 8; ++r) qkv[(mt * 16 + mb + r) * 144 + c] = (h16)acc[r];
    }
  }
  // ctx = gelu(LN(m2v @ Wm2v^T + b))  (wave 7: M=16,N=48,K=224)
  if (wave == 7) {
    v16h am[7];
    #pragma unroll
    for (int k = 0; k < 7; ++k) am[k] = frag_A(m2vA, 224, 0, k * 32);
    v8f a3[3];
    #pragma unroll
    for (int nt = 0; nt < 3; ++nt) {
      const int c = nt * 16 + cl;
      const float bv = p.m2vB[c];
      #pragma unroll
      for (int r = 0; r < 8; ++r) a3[nt][r] = bv;
      #pragma unroll
      for (int k = 0; k < 7; ++k) a3[nt] = wmma_f16(am[k], frag_B(p.m2vW, 224, nt * 16, k * 32), a3[nt]);
    }
    #pragma unroll
    for (int r = 0; r < 8; ++r) {
      float s = 0.f, q = 0.f;
      #pragma unroll
      for (int t = 0; t < 3; ++t) { const float v = a3[t][r]; s += v; q += v * v; }
      s = redhalf(s); q = redhalf(q);
      const float mu = s * (1.f/48.f);
      const float rs = rsqrtf(q * (1.f/48.f) - mu * mu + 1e-5f);
      #pragma unroll
      for (int t = 0; t < 3; ++t) {
        const int c = t * 16 + cl;
        ctxh[(mb + r) * 64 + c] = (h16)gelu_f((a3[t][r]-mu)*rs*p.m2vG[c] + p.m2vBeta[c]);
      }
    }
  }
  __syncthreads();

  // S4: self-attention core (per (elem,head) job), vectorized LDS access;
  // output o overwrites the (dead) q columns of the same head
  for (int job = wave; job < 32; job += 8) {
    const int e = job >> 1, hh = job & 1;
    const h16* base = qkv + (e * 22) * 144;
    if (lane < 22) {
      const int qc = hh * 24, kc = 48 + hh * 24, vc = 96 + hh * 24;
      const h16* bq = base + lane * 144 + qc;
      H24 qq;
      qq.u[0] = *(const v4u*)(bq); qq.u[1] = *(const v4u*)(bq + 8); qq.u[2] = *(const v4u*)(bq + 16);
      float qv[24];
      #pragma unroll
      for (int d = 0; d < 24; ++d) qv[d] = (float)qq.h[d] * 0.204124145231932f;
      float sc[22]; float mx = -3.0e38f;
      #pragma unroll
      for (int kt = 0; kt < 22; ++kt) {
        const h16* bk = base + kt * 144 + kc;
        H24 kk;
        kk.u[0] = *(const v4u*)(bk); kk.u[1] = *(const v4u*)(bk + 8); kk.u[2] = *(const v4u*)(bk + 16);
        float s = 0.f;
        #pragma unroll
        for (int d = 0; d < 24; ++d) s += qv[d] * (float)kk.h[d];
        sc[kt] = s; mx = fmaxf(mx, s);
      }
      float sum = 0.f;
      #pragma unroll
      for (int kt = 0; kt < 22; ++kt) { sc[kt] = expf(sc[kt] - mx); sum += sc[kt]; }
      const float inv = 1.f / sum;
      float ov[24];
      #pragma unroll
      for (int d = 0; d < 24; ++d) ov[d] = 0.f;
      #pragma unroll
      for (int kt = 0; kt < 22; ++kt) {
        const h16* bv = base + kt * 144 + vc;
        H24 vv;
        vv.u[0] = *(const v4u*)(bv); vv.u[1] = *(const v4u*)(bv + 8); vv.u[2] = *(const v4u*)(bv + 16);
        const float w = sc[kt] * inv;
        #pragma unroll
        for (int d = 0; d < 24; ++d) ov[d] += w * (float)vv.h[d];
      }
      H24 oo;
      #pragma unroll
      for (int d = 0; d < 24; ++d) oo.h[d] = (h16)ov[d];
      h16* bo = qkv + (e * 22 + lane) * 144 + qc;
      *(v4u*)(bo) = oo.u[0]; *(v4u*)(bo + 8) = oo.u[1]; *(v4u*)(bo + 16) = oo.u[2];
    }
  }
  __syncthreads();
  // zero cols 48..63 (K-pad for out-proj A-frags)
  for (int i = tid; i < 352 * 16; i += 256) {
    const int r = i >> 4, c = 48 + (i & 15);
    qkv[r * 144 + c] = (h16)0;
  }
  __syncthreads();

  // S5: out-proj + residual + LN -> tok
  for (int mt = wave; mt < 22; mt += 8) {
    const v16h a0 = frag_A(qkv, 144, mt * 16, 0);
    const v16h a1 = frag_A(qkv, 144, mt * 16, 32);
    v8f acc[3];
    #pragma unroll
    for (int nt = 0; nt < 3; ++nt) {
      const int c = nt * 16 + cl;
      const float bv = p.sa_outb[c];
      #pragma unroll
      for (int r = 0; r < 8; ++r) acc[nt][r] = bv;
      acc[nt] = wmma_f16(a0, frag_B(p.saoutW, 64, nt * 16, 0), acc[nt]);
      acc[nt] = wmma_f16(a1, frag_B(p.saoutW, 64, nt * 16, 32), acc[nt]);
    }
    #pragma unroll
    for (int r = 0; r < 8; ++r) {
      const int row = mt * 16 + mb + r;
      float s = 0.f, q = 0.f;
      #pragma unroll
      for (int t = 0; t < 3; ++t) {
        const float v = acc[t][r] + (float)tok[row * 64 + t * 16 + cl];
        acc[t][r] = v; s += v; q += v * v;
      }
      s = redhalf(s); q = redhalf(q);
      const float mu = s * (1.f/48.f);
      const float rs = rsqrtf(q * (1.f/48.f) - mu * mu + 1e-5f);
      #pragma unroll
      for (int t = 0; t < 3; ++t) {
        const int c = t * 16 + cl;
        tok[row * 64 + c] = (h16)((acc[t][r]-mu)*rs*p.san_g[c] + p.san_b[c]);
      }
    }
  }
  __syncthreads();

  // S6: FF1 48->96 + gelu, into qkv as [352][96]
  for (int mt = wave; mt < 22; mt += 8) {
    const v16h a0 = frag_A(tok, 64, mt * 16, 0);
    const v16h a1 = frag_A(tok, 64, mt * 16, 32);
    #pragma unroll
    for (int nt = 0; nt < 6; ++nt) {
      const int c = nt * 16 + cl;
      const float bv = p.ff_b1[c];
      v8f acc;
      #pragma unroll
      for (int r = 0; r < 8; ++r) acc[r] = bv;
      acc = wmma_f16(a0, frag_B(p.ffW1, 64, nt * 16, 0), acc);
      acc = wmma_f16(a1, frag_B(p.ffW1, 64, nt * 16, 32), acc);
      #pragma unroll
      for (int r = 0; r < 8; ++r) qkv[(mt * 16 + mb + r) * 96 + c] = (h16)gelu_f(acc[r]);
    }
  }
  __syncthreads();

  // S7: FF2 96->48 + residual + LN -> tok
  for (int mt = wave; mt < 22; mt += 8) {
    v16h a[3];
    #pragma unroll
    for (int k = 0; k < 3; ++k) a[k] = frag_A(qkv, 96, mt * 16, k * 32);
    v8f acc[3];
    #pragma unroll
    for (int nt = 0; nt < 3; ++nt) {
      const int c = nt * 16 + cl;
      const float bv = p.ff_b2[c];
      #pragma unroll
      for (int r = 0; r < 8; ++r) acc[nt][r] = bv;
      #pragma unroll
      for (int k = 0; k < 3; ++k) acc[nt] = wmma_f16(a[k], frag_B(p.ffW2, 96, nt * 16, k * 32), acc[nt]);
    }
    #pragma unroll
    for (int r = 0; r < 8; ++r) {
      const int row = mt * 16 + mb + r;
      float s = 0.f, q = 0.f;
      #pragma unroll
      for (int t = 0; t < 3; ++t) {
        const float v = acc[t][r] + (float)tok[row * 64 + t * 16 + cl];
        acc[t][r] = v; s += v; q += v * v;
      }
      s = redhalf(s); q = redhalf(q);
      const float mu = s * (1.f/48.f);
      const float rs = rsqrtf(q * (1.f/48.f) - mu * mu + 1e-5f);
      #pragma unroll
      for (int t = 0; t < 3; ++t) {
        const int c = t * 16 + cl;
        tok[row * 64 + c] = (h16)((acc[t][r]-mu)*rs*p.safn_g[c] + p.safn_b[c]);
      }
    }
  }
  __syncthreads();

  // S8a: cross-attn v = ctx @ Wv^T + bv (1 kv token => softmax == 1)
  if (wave == 0) {
    const v16h a0 = frag_A(ctxh, 64, 0, 0);
    const v16h a1 = frag_A(ctxh, 64, 0, 32);
    #pragma unroll
    for (int nt = 0; nt < 3; ++nt) {
      const int c = nt * 16 + cl;
      const float bv = p.ca_inb[96 + c];
      v8f acc;
      #pragma unroll
      for (int r = 0; r < 8; ++r) acc[r] = bv;
      acc = wmma_f16(a0, frag_B(p.cavW, 64, nt * 16, 0), acc);
      acc = wmma_f16(a1, frag_B(p.cavW, 64, nt * 16, 32), acc);
      #pragma unroll
      for (int r = 0; r < 8; ++r) vbuf[(mb + r) * 64 + c] = (h16)acc[r];
    }
  }
  __syncthreads();
  // S8b: co = v @ out_w^T + out_b (fp32 to fsc)
  if (wave == 0) {
    const v16h a0 = frag_A(vbuf, 64, 0, 0);
    const v16h a1 = frag_A(vbuf, 64, 0, 32);
    #pragma unroll
    for (int nt = 0; nt < 3; ++nt) {
      const int c = nt * 16 + cl;
      const float bv = p.ca_outb[c];
      v8f acc;
      #pragma unroll
      for (int r = 0; r < 8; ++r) acc[r] = bv;
      acc = wmma_f16(a0, frag_B(p.caoutW, 64, nt * 16, 0), acc);
      acc = wmma_f16(a1, frag_B(p.caoutW, 64, nt * 16, 32), acc);
      #pragma unroll
      for (int r = 0; r < 8; ++r) fsc[(mb + r) * 48 + c] = acc[r];
    }
  }
  __syncthreads();

  // S9: tok = LN(tok + co[elem])   (vectorized row access)
  for (int r = tid; r < 352; r += 256) {
    const int e = r / 22;
    h16* tr = tok + r * 64;
    R48 row;
    #pragma unroll
    for (int j = 0; j < 6; ++j) row.u[j] = *(const v4u*)(tr + 8 * j);
    float v[48]; float s = 0.f, q = 0.f;
    #pragma unroll
    for (int c = 0; c < 48; ++c) {
      const float x = (float)row.h[c] + fsc[e * 48 + c];
      v[c] = x; s += x; q += x * x;
    }
    const float mu = s * (1.f/48.f);
    const float rs = rsqrtf(q * (1.f/48.f) - mu * mu + 1e-5f);
    R48 outv;
    #pragma unroll
    for (int c = 0; c < 48; ++c) outv.h[c] = (h16)((v[c]-mu)*rs*p.can_g[c] + p.can_b[c]);
    #pragma unroll
    for (int j = 0; j < 6; ++j) *(v4u*)(tr + 8 * j) = outv.u[j];
  }
  __syncthreads();

  // S10: mean-pool over tokens
  for (int i = tid; i < 768; i += 256) {
    const int e = i / 48, c = i - e * 48;
    float s = 0.f;
    #pragma unroll
    for (int t = 0; t < 22; ++t) s += (float)tok[(e * 22 + t) * 64 + c];
    pooled[e * 64 + c] = (h16)(s * (1.f/22.f));
  }
  __syncthreads();

  // S11: xp = gelu(LN(pooled @ Wpool^T + b)) -> global (f16)
  if (wave == 0) {
    const v16h a0 = frag_A(pooled, 64, 0, 0);
    const v16h a1 = frag_A(pooled, 64, 0, 32);
    v8f acc[4];
    #pragma unroll
    for (int nt = 0; nt < 4; ++nt) {
      const int c = nt * 16 + cl;
      const float bv = p.pool_b[c];
      #pragma unroll
      for (int r = 0; r < 8; ++r) acc[nt][r] = bv;
      acc[nt] = wmma_f16(a0, frag_B(p.poolW, 64, nt * 16, 0), acc[nt]);
      acc[nt] = wmma_f16(a1, frag_B(p.poolW, 64, nt * 16, 32), acc[nt]);
    }
    #pragma unroll
    for (int r = 0; r < 8; ++r) {
      float s = 0.f, q = 0.f;
      #pragma unroll
      for (int t = 0; t < 4; ++t) { const float v = acc[t][r]; s += v; q += v * v; }
      s = redhalf(s); q = redhalf(q);
      const float mu = s * (1.f/64.f);
      const float rs = rsqrtf(q * (1.f/64.f) - mu * mu + 1e-5f);
      #pragma unroll
      for (int t = 0; t < 4; ++t) {
        const int c = t * 16 + cl;
        p.xp[(size_t)(e0 + mb + r) * 64 + c] =
            (h16)gelu_f((acc[t][r]-mu)*rs*p.pool_g[c] + p.pool_beta[c]);
      }
    }
  }
}

// ---------------- recurrent ACT loop ---------------------------------------
struct RecP {
  const h16 *xp, *zW1, *zW2, *yW1, *yW2, *h1Wf;
  const float *z_b1, *z_b2, *z_g, *z_beta;
  const float *y_b1, *y_b2, *y_g, *y_beta;
  const float *headW, *headB;
  const float *h1B, *h2W, *h2B;
  float* out;
};

__global__ __launch_bounds__(128)
void rec_kernel(RecP p) {
  __shared__ __align__(16) h16 wbuf[4][5120];   // per wave: tcat[16][192] + hbuf[16][128]
  __shared__ float sH1b[32], sH2[32], sHd[64];

  const int tid  = threadIdx.x;
  const int lane = tid & 31;
  const int wave = tid >> 5;
  const int cl   = lane & 15;
  const int mb   = (lane & 16) ? 8 : 0;
  h16* tcat = wbuf[wave];          // [16][192]: cols 0..63 xp | 64..127 y | 128..191 z
  h16* hbuf = wbuf[wave] + 3072;   // [16][128]: hidden (100, zero-padded)
  const int e0 = blockIdx.x * 64 + wave * 16;

  if (tid < 32) { sH1b[tid] = p.h1B[tid]; sH2[tid] = p.h2W[tid]; }
  if (tid < 64) sHd[tid] = p.headW[tid];
  const float headb = p.headB[0];
  const float h2b   = p.h2B[0];

  // init: xp into tcat, y/z/hbuf zero
  for (int i = lane; i < 1024; i += 32) {
    const int r = i >> 6, c = i & 63;
    tcat[r * 192 + c] = p.xp[(size_t)(e0 + r) * 64 + c];
  }
  for (int i = lane; i < 16 * 128; i += 32) { const int r = i >> 7, c = i & 127; tcat[r * 192 + 64 + c] = (h16)0; }
  for (int i = lane; i < 2048; i += 32) hbuf[i] = (h16)0;
  __syncthreads();

  float zs[4][8], ys[4][8];       // fp32 state in C-fragment layout
  float accH[8], accP[8];
  #pragma unroll
  for (int t = 0; t < 4; ++t)
    #pragma unroll
    for (int r = 0; r < 8; ++r) { zs[t][r] = 0.f; ys[t][r] = 0.f; }
  #pragma unroll
  for (int r = 0; r < 8; ++r) { accH[r] = 0.f; accP[r] = 0.f; }

  #pragma unroll 1
  for (int step = 0; step < 22; ++step) {
    // ---- z_up GEMM1: h = gelu([xp|y|z] @ W1^T + b1)   M16 N112 K192
    {
      v16h az[6];
      #pragma unroll
      for (int k = 0; k < 6; ++k) az[k] = frag_A(tcat, 192, 0, k * 32);
      #pragma unroll
      for (int nt = 0; nt < 7; ++nt) {
        const int c = nt * 16 + cl;
        const float bv = (c < 100) ? p.z_b1[c] : 0.f;
        v8f acc;
        #pragma unroll
        for (int r = 0; r < 8; ++r) acc[r] = bv;
        #pragma unroll
        for (int k = 0; k < 6; ++k) acc = wmma_f16(az[k], frag_B(p.zW1, 192, nt * 16, k * 32), acc);
        #pragma unroll
        for (int r = 0; r < 8; ++r) hbuf[(mb + r) * 128 + c] = (h16)gelu_f(acc[r]);
      }
    }
    __syncthreads();
    // ---- z_up GEMM2 + LN + state update
    {
      v16h ah[4];
      #pragma unroll
      for (int k = 0; k < 4; ++k) ah[k] = frag_A(hbuf, 128, 0, k * 32);
      v8f acc[4];
      #pragma unroll
      for (int nt = 0; nt < 4; ++nt) {
        const int c = nt * 16 + cl;
        const float bv = p.z_b2[c];
        #pragma unroll
        for (int r = 0; r < 8; ++r) acc[nt][r] = bv;
        #pragma unroll
        for (int k = 0; k < 4; ++k) acc[nt] = wmma_f16(ah[k], frag_B(p.zW2, 128, nt * 16, k * 32), acc[nt]);
      }
      #pragma unroll
      for (int r = 0; r < 8; ++r) {
        float s = 0.f, q = 0.f;
        #pragma unroll
        for (int t = 0; t < 4; ++t) { const float v = acc[t][r]; s += v; q += v * v; }
        s = redhalf(s); q = redhalf(q);
        const float mu = s * (1.f/64.f);
        const float rs = rsqrtf(q * (1.f/64.f) - mu * mu + 1e-5f);
        #pragma unroll
        for (int t = 0; t < 4; ++t) {
          const int c = t * 16 + cl;
          zs[t][r] += (acc[t][r]-mu)*rs*p.z_g[c] + p.z_beta[c];
          tcat[(mb + r) * 192 + 128 + c] = (h16)zs[t][r];
        }
      }
    }
    __syncthreads();
    // ---- y_up GEMM1: h = gelu([y|z] @ W1^T + b1)   M16 N112 K128
    {
      v16h ay[4];
      #pragma unroll
      for (int k = 0; k < 4; ++k) ay[k] = frag_A(tcat + 64, 192, 0, k * 32);
      #pragma unroll
      for (int nt = 0; nt < 7; ++nt) {
        const int c = nt * 16 + cl;
        const float bv = (c < 100) ? p.y_b1[c] : 0.f;
        v8f acc;
        #pragma unroll
        for (int r = 0; r < 8; ++r) acc[r] = bv;
        #pragma unroll
        for (int k = 0; k < 4; ++k) acc = wmma_f16(ay[k], frag_B(p.yW1, 128, nt * 16, k * 32), acc);
        #pragma unroll
        for (int r = 0; r < 8; ++r) hbuf[(mb + r) * 128 + c] = (h16)gelu_f(acc[r]);
      }
    }
    __syncthreads();
    // ---- y_up GEMM2 + LN + state update
    {
      v16h ah[4];
      #pragma unroll
      for (int k = 0; k < 4; ++k) ah[k] = frag_A(hbuf, 128, 0, k * 32);
      v8f acc[4];
      #pragma unroll
      for (int nt = 0; nt < 4; ++nt) {
        const int c = nt * 16 + cl;
        const float bv = p.y_b2[c];
        #pragma unroll
        for (int r = 0; r < 8; ++r) acc[nt][r] = bv;
        #pragma unroll
        for (int k = 0; k < 4; ++k) acc[nt] = wmma_f16(ah[k], frag_B(p.yW2, 128, nt * 16, k * 32), acc[nt]);
      }
      #pragma unroll
      for (int r = 0; r < 8; ++r) {
        float s = 0.f, q = 0.f;
        #pragma unroll
        for (int t = 0; t < 4; ++t) { const float v = acc[t][r]; s += v; q += v * v; }
        s = redhalf(s); q = redhalf(q);
        const float mu = s * (1.f/64.f);
        const float rs = rsqrtf(q * (1.f/64.f) - mu * mu + 1e-5f);
        #pragma unroll
        for (int t = 0; t < 4; ++t) {
          const int c = t * 16 + cl;
          ys[t][r] += (acc[t][r]-mu)*rs*p.y_g[c] + p.y_beta[c];
          tcat[(mb + r) * 192 + 64 + c] = (h16)ys[t][r];
        }
      }
    }
    __syncthreads();

    // ---- head + halt (only steps >= MIN_STEPS contribute: h==0 earlier)
    if (step >= 12) {
      // pred = y @ headW + b   (dot in C-fragment layout + half-wave reduce)
      float pr[8];
      #pragma unroll
      for (int r = 0; r < 8; ++r) {
        float s = 0.f;
        #pragma unroll
        for (int t = 0; t < 4; ++t) s += ys[t][r] * sHd[t * 16 + cl];
        s = redhalf(s);
        pr[r] = s + headb;
      }
      // halt hidden = gelu(y @ W1h^T + b1): WMMA M16 N32 K64, A = y in tcat
      const v16h ha0 = frag_A(tcat + 64, 192, 0, 0);
      const v16h ha1 = frag_A(tcat + 64, 192, 0, 32);
      v8f hac[2];
      #pragma unroll
      for (int ht = 0; ht < 2; ++ht) {
        const int c = ht * 16 + cl;
        const float bv = sH1b[c];
        #pragma unroll
        for (int r = 0; r < 8; ++r) hac[ht][r] = bv;
        hac[ht] = wmma_f16(ha0, frag_B(p.h1Wf, 64, ht * 16, 0), hac[ht]);
        hac[ht] = wmma_f16(ha1, frag_B(p.h1Wf, 64, ht * 16, 32), hac[ht]);
      }
      // logit = gelu(hidden) . W2 + b2 ; sigmoid ; accumulate
      #pragma unroll
      for (int r = 0; r < 8; ++r) {
        float pj = gelu_f(hac[0][r]) * sH2[cl] + gelu_f(hac[1][r]) * sH2[16 + cl];
        pj = redhalf(pj);
        const float hh = 1.f / (1.f + expf(-(pj + h2b)));
        accH[r] += hh;
        accP[r] += hh * pr[r];
      }
    }
  }

  if (lane == 0) {
    #pragma unroll
    for (int r = 0; r < 8; ++r) p.out[e0 + r] = accP[r] / (accH[r] + 1e-8f);
  } else if (lane == 16) {
    #pragma unroll
    for (int r = 0; r < 8; ++r) p.out[e0 + 8 + r] = accP[r] / (accH[r] + 1e-8f);
  }
}

// ---------------- host launcher --------------------------------------------
extern "C" void kernel_launch(void* const* d_in, const int* in_sizes, int n_in,
                              void* d_out, int out_size, void* d_ws, size_t ws_size,
                              hipStream_t stream) {
  const float* F[64];
  for (int i = 0; i < n_in && i < 64; ++i) F[i] = (const float*)d_in[i];
  h16* ws = (h16*)d_ws;
  const int nB = in_sizes[0] / 332;   // 65536

  struct Cvt { int srcIdx, N, K, Np, Kp, rowOff, srcLd; size_t off; };
  const Cvt cv[13] = {
    {5,  48, 200,  48, 224,  0, 200, OFF_M2VW},
    {9, 144,  48, 144,  64,  0,  48, OFF_SAINW},
    {11, 48,  48,  48,  64,  0,  48, OFF_SAOUTW},
    {15, 96,  48,  96,  64,  0,  48, OFF_FFW1},
    {17, 48,  96,  48,  96,  0,  96, OFF_FFW2},
    {21, 48,  48,  48,  64, 96,  48, OFF_CAVW},   // v-rows of ca.in_w
    {23, 48,  48,  48,  64,  0,  48, OFF_CAOUTW},
    {27, 64,  48,  64,  64,  0,  48, OFF_POOLW},
    {31, 100, 192, 112, 192, 0, 192, OFF_ZW1},
    {33, 64, 100,  64, 128,  0, 100, OFF_ZW2},
    {37, 100, 128, 112, 128, 0, 128, OFF_YW1},
    {39, 64, 100,  64, 128,  0, 100, OFF_YW2},
    {45, 32,  64,  32,  64,  0,  64, OFF_H1W},    // halt.W1 -> f16 (exact)
  };
  for (int i = 0; i < 13; ++i) {
    const int tot = cv[i].Np * cv[i].Kp;
    cvt_pad<<<(tot + 255) / 256, 256, 0, stream>>>(
        F[cv[i].srcIdx], ws + cv[i].off,
        cv[i].N, cv[i].K, cv[i].Np, cv[i].Kp, cv[i].rowOff, cv[i].srcLd);
  }

  EncP ep;
  ep.x = F[0];
  ep.tokW = F[1];  ep.tokB = F[2];  ep.tokG = F[3];  ep.tokBeta = F[4];
  ep.m2vB = F[6];  ep.m2vG = F[7];  ep.m2vBeta = F[8];
  ep.sa_inb = F[10]; ep.sa_outb = F[12]; ep.san_g = F[13]; ep.san_b = F[14];
  ep.ff_b1 = F[16]; ep.ff_b2 = F[18]; ep.safn_g = F[19]; ep.safn_b = F[20];
  ep.ca_inb = F[22]; ep.ca_outb = F[24]; ep.can_g = F[25]; ep.can_b = F[26];
  ep.pool_b = F[28]; ep.pool_g = F[29]; ep.pool_beta = F[30];
  ep.m2vW  = ws + OFF_M2VW;  ep.sainW = ws + OFF_SAINW; ep.saoutW = ws + OFF_SAOUTW;
  ep.ffW1  = ws + OFF_FFW1;  ep.ffW2  = ws + OFF_FFW2;  ep.cavW   = ws + OFF_CAVW;
  ep.caoutW = ws + OFF_CAOUTW; ep.poolW = ws + OFF_POOLW;
  ep.xp = ws + OFF_XP;
  enc_kernel<<<nB / 16, 256, ENC_LDS_BYTES, stream>>>(ep);

  RecP rp;
  rp.xp = ws + OFF_XP;
  rp.zW1 = ws + OFF_ZW1; rp.zW2 = ws + OFF_ZW2;
  rp.yW1 = ws + OFF_YW1; rp.yW2 = ws + OFF_YW2;
  rp.h1Wf = ws + OFF_H1W;
  rp.z_b1 = F[32]; rp.z_b2 = F[34]; rp.z_g = F[35]; rp.z_beta = F[36];
  rp.y_b1 = F[38]; rp.y_b2 = F[40]; rp.y_g = F[41]; rp.y_beta = F[42];
  rp.headW = F[43]; rp.headB = F[44];
  rp.h1B = F[46]; rp.h2W = F[47]; rp.h2B = F[48];
  rp.out = (float*)d_out;
  rec_kernel<<<nB / 64, 128, 0, stream>>>(rp);
}
